// VisionAttention_39092792328509
// MI455X (gfx1250) — matrix operand order repaired and checked by reading, not compiled, not run on hardware
//
#include <hip/hip_runtime.h>
#include <math.h>

// ---------------- problem constants ----------------
#define T_TOK   8192
#define E_DIM   1280
#define E3      3840
#define H_HEADS 16
#define D_HEAD  80
#define DPAD    96          // D padded to 3*32 for K-dim of QK^T WMMA
#define SEG     1024
#define NSEG    8
#define SCALE_QK 0.11180339887498948f  // 1/sqrt(80)

typedef __bf16 bf16;
typedef __attribute__((ext_vector_type(16))) __bf16 v16bf;
typedef __attribute__((ext_vector_type(8)))  __bf16 v8bf;
typedef __attribute__((ext_vector_type(4)))  __bf16 v4bf;
typedef __attribute__((ext_vector_type(8)))  float  v8f;

union FragU { v16bf f; v8bf h[2]; };

__device__ inline v8f v8f_zero() {
  v8f z = {0.f, 0.f, 0.f, 0.f, 0.f, 0.f, 0.f, 0.f};
  return z;
}

// A fragment (16x32 bf16, row-major src, leading dim ld in elements).
// ISA 7.12.2: lane<16 -> M=lane, halves[0..7]=K0..7, halves[8..15]=K16..23
//             lane>=16 -> M=lane-16, halves[0..7]=K8..15, halves[8..15]=K24..31
__device__ inline v16bf load_frag_a(const bf16* base, int ld) {
  const int lane = threadIdx.x & 31;
  const int row  = lane & 15;
  const int koff = (lane & 16) ? 8 : 0;
  const bf16* p = base + (size_t)row * ld + koff;
  FragU u;
  u.h[0] = *(const v8bf*)p;
  u.h[1] = *(const v8bf*)(p + 16);
  return u.f;
}

// B fragment (32x16 bf16). Source is laid out with the N index selecting a row
// of 'base' (i.e. base[n*ld + k] == B[k][n]), so each lane loads two 16-byte
// runs of its row. lane<16 -> N=lane, K=0..7,16..23 ; lane>=16 -> N=lane-16, K=8..15,24..31.
__device__ inline v16bf load_frag_b(const bf16* base, int ld) {
  const int lane = threadIdx.x & 31;
  const int col  = lane & 15;
  const int koff = (lane & 16) ? 8 : 0;
  const bf16* p = base + (size_t)col * ld + koff;
  FragU u;
  u.h[0] = *(const v8bf*)p;
  u.h[1] = *(const v8bf*)(p + 16);
  return u.f;
}

// ---------------- f32 -> bf16 convert (4 elems/thread, packed cvt) ----------------
__global__ void va_cvt_bf16_kernel(const float* __restrict__ src,
                                   bf16* __restrict__ dst, int n4) {
  int i = blockIdx.x * blockDim.x + threadIdx.x;
  if (i >= n4) return;
  const float4 v = ((const float4*)src)[i];
  v4bf o = {(__bf16)v.x, (__bf16)v.y, (__bf16)v.z, (__bf16)v.w};
  ((v4bf*)dst)[i] = o;
}

// ---------------- WMMA GEMM: C[M,N] = A[M,K] * B[N,K]^T + bias ----------------
// A, B bf16 row-major (both contract over K). 256 threads = 8 waves (2x4),
// block tile 128x128, wave tile 64x32.
template <bool OUT_BF16>
__global__ void va_wmma_gemm_bias_kernel(const bf16* __restrict__ A,
                                         const bf16* __restrict__ B,
                                         const float* __restrict__ bias,
                                         void* __restrict__ Cout,
                                         int M, int N, int K) {
  const int wave = threadIdx.x >> 5;
  const int lane = threadIdx.x & 31;
  const int wm = wave >> 2;            // 0..1
  const int wn = wave & 3;             // 0..3
  const int row0 = blockIdx.x * 128 + wm * 64;
  const int col0 = blockIdx.y * 128 + wn * 32;

  v8f c[4][2];
#pragma unroll
  for (int i = 0; i < 4; ++i)
#pragma unroll
    for (int j = 0; j < 2; ++j) c[i][j] = v8f_zero();

  for (int k0 = 0; k0 < K; k0 += 32) {
    if (k0 + 32 < K) {  // L2-resident streams; hint next K panel
      __builtin_prefetch(A + (size_t)row0 * K + k0 + 32, 0, 1);
      __builtin_prefetch(B + (size_t)col0 * K + k0 + 32, 0, 1);
    }
    v16bf a[4], b[2];
#pragma unroll
    for (int i = 0; i < 4; ++i)
      a[i] = load_frag_a(A + (size_t)(row0 + i * 16) * K + k0, K);
#pragma unroll
    for (int j = 0; j < 2; ++j)
      b[j] = load_frag_b(B + (size_t)(col0 + j * 16) * K + k0, K);
#pragma unroll
    for (int i = 0; i < 4; ++i)
#pragma unroll
      for (int j = 0; j < 2; ++j)
        c[i][j] = __builtin_amdgcn_wmma_f32_16x16x32_bf16(
            false, a[i], false, b[j], (short)0, c[i][j], false, false);
  }

  const int col = lane & 15;
  const int rb  = (lane & 16) ? 8 : 0;
#pragma unroll
  for (int j = 0; j < 2; ++j) {
    const int cc = col0 + j * 16 + col;
    const float bv = bias[cc];
#pragma unroll
    for (int i = 0; i < 4; ++i) {
#pragma unroll
      for (int r = 0; r < 8; ++r) {
        const int rr = row0 + i * 16 + rb + r;
        const float v = c[i][j][r] + bv;
        if (OUT_BF16)
          ((bf16*)Cout)[(size_t)rr * N + cc] = (__bf16)v;
        else
          ((float*)Cout)[(size_t)rr * N + cc] = v;
      }
    }
  }
}

// ---------------- RoPE + scatter into attention layouts ----------------
// qkv: bf16 [T, 3E]. Writes:
//   Qp, Kp : bf16 [NSEG, H, SEG, DPAD]  (RoPE applied, cols 80..95 zero)
//   Vt     : bf16 [NSEG, H, D_HEAD, SEG] (transposed)
__global__ void va_rope_scatter_kernel(const bf16* __restrict__ qkv,
                                       const float* __restrict__ cosb,
                                       const float* __restrict__ sinb,
                                       bf16* __restrict__ Qp,
                                       bf16* __restrict__ Kp,
                                       bf16* __restrict__ Vt) {
  int idx = blockIdx.x * blockDim.x + threadIdx.x;
  if (idx >= T_TOK * H_HEADS * D_HEAD) return;
  const int d = idx % D_HEAD;
  const int h = (idx / D_HEAD) % H_HEADS;
  const int t = idx / (D_HEAD * H_HEADS);

  const float c = cosb[t * D_HEAD + d];
  const float s = sinb[t * D_HEAD + d];
  const size_t base = (size_t)t * E3 + h * D_HEAD;
  const float qv = (float)qkv[base + d];
  const float kv = (float)qkv[base + E_DIM + d];
  const float vv = (float)qkv[base + 2 * E_DIM + d];
  const int  dro = (d < D_HEAD / 2) ? d + D_HEAD / 2 : d - D_HEAD / 2;
  const float sg = (d < D_HEAD / 2) ? -1.f : 1.f;
  const float qr = (float)qkv[base + dro];
  const float kr = (float)qkv[base + E_DIM + dro];
  const float qo = qv * c + sg * qr * s;
  const float ko = kv * c + sg * kr * s;

  const int n = t >> 10;
  const int l = t & (SEG - 1);
  const size_t rowbase = ((size_t)(n * H_HEADS + h) * SEG + l) * DPAD;
  Qp[rowbase + d] = (__bf16)qo;
  Kp[rowbase + d] = (__bf16)ko;
  if (d < DPAD - D_HEAD) {            // zero the K-dim padding
    Qp[rowbase + D_HEAD + d] = (__bf16)0.f;
    Kp[rowbase + D_HEAD + d] = (__bf16)0.f;
  }
  Vt[((size_t)(n * H_HEADS + h) * D_HEAD + d) * SEG + l] = (__bf16)vv;
}

// ---------------- block-diagonal attention ----------------
// grid = NSEG*H*(SEG/64) blocks of 128 threads (4 waves); wave owns 16 q rows.
// LDS per wave: bf16 scores/P [16][1024] (32 KB, written in place).
// 4 waves -> 128.75 KB/block -> 2 blocks per 320 KB WGP (8 waves resident).
#define SMEM_RED_OFF (4 * 16 * 1024 * 2)            // 131072
#define SMEM_INV_OFF (SMEM_RED_OFF + 4 * 32 * 4)    // +512
#define SMEM_BYTES   (SMEM_INV_OFF + 4 * 16 * 4)    // +256 = 131840

__global__ void va_attn_kernel(const bf16* __restrict__ Qp,
                               const bf16* __restrict__ Kp,
                               const bf16* __restrict__ Vt,
                               bf16* __restrict__ attnb) {
  extern __shared__ __align__(16) char smem[];
  const int wave = threadIdx.x >> 5;                // 0..3
  const int lane = threadIdx.x & 31;

  const int qb = blockIdx.x & 15;                   // 16 q-blocks of 64 rows
  const int h  = (blockIdx.x >> 4) & (H_HEADS - 1);
  const int n  = blockIdx.x >> 8;
  const int qrow0 = qb * 64 + wave * 16;

  bf16*  myS   = (bf16*)smem + wave * 16 * 1024;
  float* myred = (float*)(smem + SMEM_RED_OFF) + wave * 32;
  float* myinv = (float*)(smem + SMEM_INV_OFF) + wave * 16;

  const size_t head = (size_t)(n * H_HEADS + h);
  const bf16* Qb = Qp + (head * SEG + qrow0) * DPAD;
  const bf16* Kb = Kp + head * SEG * DPAD;
  const bf16* Vb = Vt + head * D_HEAD * SEG;

  // ---- S = (Q K^T) * scale, stored bf16 into wave-private LDS ----
  v16bf aq[3];
#pragma unroll
  for (int kk = 0; kk < 3; ++kk) aq[kk] = load_frag_a(Qb + kk * 32, DPAD);

  const int col = lane & 15;
  const int rb  = (lane & 16) ? 8 : 0;
  for (int mb = 0; mb < SEG / 16; ++mb) {
    v8f c = v8f_zero();
#pragma unroll
    for (int kk = 0; kk < 3; ++kk) {
      v16bf bk = load_frag_b(Kb + (size_t)(mb * 16) * DPAD + kk * 32, DPAD);
      c = __builtin_amdgcn_wmma_f32_16x16x32_bf16(
          false, aq[kk], false, bk, (short)0, c, false, false);
    }
#pragma unroll
    for (int r = 0; r < 8; ++r)
      myS[(rb + r) * 1024 + mb * 16 + col] = (__bf16)(c[r] * SCALE_QK);
  }
  __builtin_amdgcn_wave_barrier();

  // ---- softmax: lane owns half a row; vectorized 8-wide via b128 LDS ops ----
  const int row = lane & 15;
  const int hv  = lane >> 4;
  bf16* rowp = myS + row * 1024 + hv * 512;

  float m = -3.402823466e38f;
  for (int i = 0; i < 512; i += 8) {
    const v8bf v = *(const v8bf*)(rowp + i);
#pragma unroll
    for (int j = 0; j < 8; ++j) m = fmaxf(m, (float)v[j]);
  }
  myred[hv * 16 + row] = m;
  __builtin_amdgcn_wave_barrier();
  const float M = fmaxf(myred[row], myred[16 + row]);

  float sum = 0.f;
  for (int i = 0; i < 512; i += 8) {
    const v8bf v = *(const v8bf*)(rowp + i);
    v8bf e;
#pragma unroll
    for (int j = 0; j < 8; ++j) {
      const float ev = __expf((float)v[j] - M);
      sum += ev;
      e[j] = (__bf16)ev;             // unnormalized P, written back in place
    }
    *(v8bf*)(rowp + i) = e;
  }
  myred[hv * 16 + row] = sum;
  __builtin_amdgcn_wave_barrier();
  if (hv == 0) myinv[row] = 1.f / (myred[row] + myred[16 + row]);
  __builtin_amdgcn_wave_barrier();   // re-converged: EXEC all-ones for WMMA below

  // ---- O = P @ V (unnormalized); normalization fused into the epilogue ----
  v8f o[5];
#pragma unroll
  for (int j = 0; j < 5; ++j) o[j] = v8f_zero();

  for (int mb = 0; mb < SEG / 32; ++mb) {
    v16bf ap = load_frag_a(myS + mb * 32, 1024);
#pragma unroll
    for (int j = 0; j < 5; ++j) {
      v16bf bv = load_frag_b(Vb + (size_t)(j * 16) * SEG + mb * 32, SEG);
      o[j] = __builtin_amdgcn_wmma_f32_16x16x32_bf16(
          false, ap, false, bv, (short)0, o[j], false, false);
    }
  }

  float invr[8];
#pragma unroll
  for (int r = 0; r < 8; ++r) invr[r] = myinv[rb + r];

#pragma unroll
  for (int j = 0; j < 5; ++j) {
#pragma unroll
    for (int r = 0; r < 8; ++r) {
      const int l = qrow0 + rb + r;
      const int t = n * SEG + l;
      const int e = h * D_HEAD + j * 16 + col;
      attnb[(size_t)t * E_DIM + e] = (__bf16)(o[j][r] * invr[r]);
    }
  }
}

// ---------------- host side ----------------
extern "C" void kernel_launch(void* const* d_in, const int* in_sizes, int n_in,
                              void* d_out, int out_size, void* d_ws, size_t ws_size,
                              hipStream_t stream) {
  (void)in_sizes; (void)n_in; (void)out_size; (void)ws_size;
  const float* x      = (const float*)d_in[0];
  const float* cosb   = (const float*)d_in[2];
  const float* sinb   = (const float*)d_in[3];
  const float* w_qkv  = (const float*)d_in[4];
  const float* b_qkv  = (const float*)d_in[5];
  const float* w_proj = (const float*)d_in[6];
  const float* b_proj = (const float*)d_in[7];
  float* out = (float*)d_out;

  char* ws = (char*)d_ws;
  size_t off = 0;
  auto alloc = [&](size_t bytes) -> char* {
    char* p = ws + off;
    off += (bytes + 255) & ~(size_t)255;
    return p;
  };
  bf16* xb     = (bf16*)alloc((size_t)T_TOK * E_DIM * 2);
  bf16* wqkvb  = (bf16*)alloc((size_t)E3 * E_DIM * 2);
  bf16* wprojb = (bf16*)alloc((size_t)E_DIM * E_DIM * 2);
  bf16* qkvb   = (bf16*)alloc((size_t)T_TOK * E3 * 2);
  bf16* Qp     = (bf16*)alloc((size_t)NSEG * H_HEADS * SEG * DPAD * 2);
  bf16* Kp     = (bf16*)alloc((size_t)NSEG * H_HEADS * SEG * DPAD * 2);
  bf16* Vt     = (bf16*)alloc((size_t)NSEG * H_HEADS * D_HEAD * SEG * 2);
  bf16* attnb  = (bf16*)alloc((size_t)T_TOK * E_DIM * 2);

  int n4;
  n4 = T_TOK * E_DIM / 4;
  va_cvt_bf16_kernel<<<(n4 + 255) / 256, 256, 0, stream>>>(x, xb, n4);
  n4 = E3 * E_DIM / 4;
  va_cvt_bf16_kernel<<<(n4 + 255) / 256, 256, 0, stream>>>(w_qkv, wqkvb, n4);
  n4 = E_DIM * E_DIM / 4;
  va_cvt_bf16_kernel<<<(n4 + 255) / 256, 256, 0, stream>>>(w_proj, wprojb, n4);

  va_wmma_gemm_bias_kernel<true>
      <<<dim3(T_TOK / 128, E3 / 128), 256, 0, stream>>>(
          xb, wqkvb, b_qkv, qkvb, T_TOK, E3, E_DIM);

  const int nrope = T_TOK * H_HEADS * D_HEAD;
  va_rope_scatter_kernel<<<(nrope + 255) / 256, 256, 0, stream>>>(
      qkvb, cosb, sinb, Qp, Kp, Vt);

  va_attn_kernel<<<NSEG * H_HEADS * (SEG / 64), 128, SMEM_BYTES, stream>>>(
      Qp, Kp, Vt, attnb);

  va_wmma_gemm_bias_kernel<false>
      <<<dim3(T_TOK / 128, E_DIM / 128), 256, 0, stream>>>(
          attnb, wprojb, b_proj, out, T_TOK, E_DIM, E_DIM);
}
